// SANet_29789893165548
// MI455X (gfx1250) — matrix operand order repaired and checked by reading, not compiled, not run on hardware
//
#include <hip/hip_runtime.h>
#include <hip/hip_bf16.h>

// SANet slot attention for MI455X (gfx1250, wave32, WMMA).
// Bandwidth-bound (~700MB total traffic @ 23.3TB/s); k/v/q stored f16 in
// workspace, all GEMMs via v_wmma_f32_16x16x32_f16 with f32 accumulation.
// v-tiles staged with GLOBAL_LOAD_ASYNC_TO_LDS_B128 (ASYNCcnt) in k_update.

#define DIMD 64
#define NSLOT 6
#define NITER 3
#define HIDD 128
#define BBATCH 128
#define NPIX 4096
#define LN_EPSF 1e-5f
#define EPSF 1e-8f
#define ATT_SCALE 0.125f   // 64^-0.5

typedef _Float16 h8   __attribute__((ext_vector_type(8)));
typedef _Float16 v16h __attribute__((ext_vector_type(16)));
typedef float    v8f  __attribute__((ext_vector_type(8)));
typedef float    f4   __attribute__((ext_vector_type(4)));

static __device__ __forceinline__ v8f wmma_f16(v16h a, v16h b, v8f c) {
  // (neg_a, A, neg_b, B, c_mod, C, reuse_a, reuse_b)
  return __builtin_amdgcn_wmma_f32_16x16x32_f16(false, a, false, b, (short)0, c, false, false);
}
static __device__ __forceinline__ v16h cat16(h8 lo, h8 hi) {
  return __builtin_shufflevector(lo, hi, 0,1,2,3,4,5,6,7,8,9,10,11,12,13,14,15);
}

// ---------------- slots = mu + noise * exp(sigma) -------------------------
__global__ void k_slot_init(const float* __restrict__ mu, const float* __restrict__ sigma,
                            const float* __restrict__ noise, float* __restrict__ slots) {
  int idx = blockIdx.x * 256 + threadIdx.x;
  if (idx < BBATCH * NSLOT * DIMD) {
    int d = idx & 63;
    slots[idx] = mu[d] + noise[idx] * __expf(sigma[d]);
  }
}

// ---------------- fused LN + K/V projection (WMMA) ------------------------
// block = 256 thr = 8 waves; wave handles 16 rows; block = 128 rows.
__global__ void __launch_bounds__(256) k_ln_kv(
    const float* __restrict__ x, const float* __restrict__ Wk,
    const float* __restrict__ Wv, const float* __restrict__ g,
    const float* __restrict__ bta, _Float16* __restrict__ kf,
    _Float16* __restrict__ vf) {
  __shared__ __align__(16) _Float16 Wl[2][DIMD][DIMD];   // [mat][n][k] (transposed)
  __shared__ __align__(16) _Float16 Al[8][16][DIMD];     // LN'd f16 row tiles
  int tid = threadIdx.x;
  for (int idx = tid; idx < 2 * DIMD * DIMD; idx += 256) {
    int m = idx >> 12, r = idx & 4095;
    int n = r >> 6, k = r & 63;
    const float* W = m ? Wv : Wk;
    Wl[m][n][k] = (_Float16)W[k * DIMD + n];
  }
  int w = tid >> 5, lane = tid & 31;
  int row0 = blockIdx.x * 128 + w * 16;
  if (lane < 16) {
    const float* xr = x + (size_t)(row0 + lane) * DIMD;
    float s = 0.f, s2 = 0.f;
    for (int k = 0; k < DIMD; k += 4) {
      f4 t = *(const f4*)(xr + k);
      s  += t.x + t.y + t.z + t.w;
      s2 += t.x * t.x + t.y * t.y + t.z * t.z + t.w * t.w;
    }
    float mean = s * (1.f / DIMD);
    float var  = s2 * (1.f / DIMD) - mean * mean;
    float inv  = rsqrtf(var + LN_EPSF);
    for (int k = 0; k < DIMD; ++k)
      Al[w][lane][k] = (_Float16)((xr[k] - mean) * inv * g[k] + bta[k]);
  }
  __syncthreads();

  int hi = lane >> 4, ln16 = lane & 15;
  v16h afr[2];
#pragma unroll
  for (int c = 0; c < 2; ++c) {
    h8 r1 = *(const h8*)&Al[w][ln16][c * 32 + hi * 8];
    h8 r2 = *(const h8*)&Al[w][ln16][c * 32 + 16 + hi * 8];
    afr[c] = cat16(r1, r2);
  }
#pragma unroll
  for (int m = 0; m < 2; ++m) {
    _Float16* out = m ? vf : kf;
#pragma unroll
    for (int t = 0; t < 4; ++t) {
      v8f acc = {};
#pragma unroll
      for (int c = 0; c < 2; ++c) {
        int nn2 = t * 16 + ln16;
        h8 b1v = *(const h8*)&Wl[m][nn2][c * 32 + hi * 8];
        h8 b2v = *(const h8*)&Wl[m][nn2][c * 32 + 16 + hi * 8];
        acc = wmma_f16(afr[c], cat16(b1v, b2v), acc);
      }
#pragma unroll
      for (int r = 0; r < 8; ++r)
        out[(size_t)(row0 + r + 8 * hi) * DIMD + t * 16 + ln16] = (_Float16)acc[r];
    }
  }
}

// ---------------- q = LN(slots) @ Wq, padded to 16 rows, f16 --------------
__global__ void __launch_bounds__(256) k_qproj(
    const float* __restrict__ slots, const float* __restrict__ Wq,
    const float* __restrict__ g, const float* __restrict__ bta,
    _Float16* __restrict__ qf) {
  __shared__ float sl[NSLOT][DIMD];
  __shared__ float lns[NSLOT][DIMD];
  __shared__ float m6[NSLOT], iv6[NSLOT];
  int b = blockIdx.x, tid = threadIdx.x;
  for (int idx = tid; idx < NSLOT * DIMD; idx += 256)
    sl[idx >> 6][idx & 63] = slots[b * NSLOT * DIMD + idx];
  __syncthreads();
  if (tid < NSLOT) {
    float s = 0.f, s2 = 0.f;
    for (int k = 0; k < DIMD; ++k) { float v = sl[tid][k]; s += v; s2 += v * v; }
    float mean = s / DIMD, var = s2 / DIMD - mean * mean;
    m6[tid] = mean; iv6[tid] = rsqrtf(var + LN_EPSF);
  }
  __syncthreads();
  for (int idx = tid; idx < NSLOT * DIMD; idx += 256) {
    int s = idx >> 6, d = idx & 63;
    lns[s][d] = (sl[s][d] - m6[s]) * iv6[s] * g[d] + bta[d];
  }
  __syncthreads();
  for (int idx = tid; idx < 16 * DIMD; idx += 256) {
    int s = idx >> 6, d = idx & 63;
    float acc = 0.f;
    if (s < NSLOT)
      for (int k = 0; k < DIMD; ++k) acc += lns[s][k] * Wq[k * DIMD + d];
    qf[(size_t)(b * 16 + s) * DIMD + d] = (_Float16)acc;
  }
}

// ---------------- dots + slot-axis softmax (WMMA) -------------------------
// grid = B*32 blocks; block covers 128 pixels, one 16-pixel tile per wave.
__global__ void __launch_bounds__(256) k_dots(
    const _Float16* __restrict__ kf, const _Float16* __restrict__ qf,
    float* __restrict__ attn_out, float* __restrict__ spart, int it) {
  __shared__ float sred[8][16][NSLOT];
  int blk = blockIdx.x;
  int b = blk >> 5, nb = blk & 31;
  int tid = threadIdx.x, w = tid >> 5, lane = tid & 31;
  int hi = lane >> 4, ln16 = lane & 15;
  int p0 = nb * 128 + w * 16;
  const _Float16* qb = qf + (size_t)b * 16 * DIMD;
  const _Float16* kb = kf + ((size_t)b * NPIX + p0) * DIMD;
  v8f acc = {};
#pragma unroll
  for (int c = 0; c < 2; ++c) {
    h8 a1  = *(const h8*)(qb + ln16 * DIMD + c * 32 + hi * 8);
    h8 a2  = *(const h8*)(qb + ln16 * DIMD + c * 32 + 16 + hi * 8);
    h8 b1v = *(const h8*)(kb + ln16 * DIMD + c * 32 + hi * 8);
    h8 b2v = *(const h8*)(kb + ln16 * DIMD + c * 32 + 16 + hi * 8);
    acc = wmma_f16(cat16(a1, a2), cat16(b1v, b2v), acc);
  }
  if (hi == 0) {   // rows (slots) 0..5 live in acc[0..5] of lanes 0..15
    float vv[NSLOT];
    float vmax = -1e30f;
#pragma unroll
    for (int r = 0; r < NSLOT; ++r) { vv[r] = acc[r] * ATT_SCALE; vmax = fmaxf(vmax, vv[r]); }
    float ssum = 0.f;
#pragma unroll
    for (int r = 0; r < NSLOT; ++r) { vv[r] = __expf(vv[r] - vmax); ssum += vv[r]; }
    float inv = 1.f / ssum;
    int p = p0 + ln16;
    float* ao = attn_out + (size_t)(b * NITER + it) * NSLOT * NPIX + p;
#pragma unroll
    for (int r = 0; r < NSLOT; ++r) {
      float a = vv[r] * inv + EPSF;
      ao[(size_t)r * NPIX] = a;
      sred[w][ln16][r] = a;
    }
  }
  __syncthreads();
  if (tid < NSLOT) {     // deterministic block partial of attn row-sum
    float s = 0.f;
    for (int ww = 0; ww < 8; ++ww)
      for (int l = 0; l < 16; ++l) s += sred[ww][l][tid];
    spart[((size_t)b * 32 + nb) * 8 + tid] = s;
  }
}

// ---------------- updates (WMMA over pixels-K) + rel/GRU/MLP --------------
__global__ void __launch_bounds__(256) k_update(
    const _Float16* __restrict__ vf, const float* __restrict__ attn_out,
    const float* __restrict__ spart, float* __restrict__ slots,
    const float* __restrict__ gWih, const float* __restrict__ gWhh,
    const float* __restrict__ gbih, const float* __restrict__ gbhh,
    const float* __restrict__ oW1, const float* __restrict__ ob1,
    const float* __restrict__ oW2, const float* __restrict__ ob2,
    const float* __restrict__ rW1, const float* __restrict__ rb1,
    const float* __restrict__ rW2, const float* __restrict__ rb2,
    const float* __restrict__ lng, const float* __restrict__ lnb,
    float* __restrict__ out_slots, int it) {
  __shared__ __align__(16) _Float16 vt[8][32 * DIMD];  // per-wave v tile (32KB)
  __shared__ float up[8][NSLOT][DIMD];                 // per-wave partial C (12KB)
  __shared__ float updf[NSLOT][DIMD];
  __shared__ float prevs[NSLOT][DIMD];
  __shared__ float relv[NSLOT][DIMD];
  __shared__ float snew[NSLOT][DIMD];
  __shared__ float hidv[HIDD];
  __shared__ float giA[NSLOT][3 * DIMD];
  __shared__ float ghA[NSLOT][3 * DIMD];
  __shared__ float mrow[NSLOT], ivrow[NSLOT];

  int b = blockIdx.x, tid = threadIdx.x, w = tid >> 5, lane = tid & 31;
  int hi = lane >> 4, ln16 = lane & 15;
  const float* attb = attn_out + (size_t)(b * NITER + it) * NSLOT * NPIX;

  // ---- phase 1: updates_raw = attn @ v, K = 4096 pixels, wave-split -----
  unsigned vtbase = (unsigned)(uintptr_t)&vt[w][0];
  v8f acc[4] = {};
  for (int s = 0; s < 16; ++s) {
    int pk = (w + s * 8) * 32;                     // 32-pixel K chunk
    const _Float16* vsrc = vf + ((size_t)b * NPIX + pk) * DIMD;
    // WAR: previous chunk's LDS fragment reads must retire before overwrite
    asm volatile("s_wait_dscnt 0x0" ::: "memory");
    // stage 32x64 f16 tile straight into LDS (ASYNCcnt path, no VGPR bounce)
#pragma unroll
    for (int j = 0; j < 8; ++j) {
      unsigned lds = vtbase + (unsigned)((j * 32 + lane) * 16);
      unsigned long long ga =
          (unsigned long long)(uintptr_t)(vsrc + (j * 32 + lane) * 8);
      asm volatile("global_load_async_to_lds_b128 %0, %1, off"
                   :: "v"(lds), "v"(ga) : "memory");
    }
    // prefetch the next chunk's tile into cache while async copies land
    if (s < 15) {
      const _Float16* vnext = vf + ((size_t)b * NPIX + (w + (s + 1) * 8) * 32) * DIMD;
      __builtin_prefetch(vnext + (size_t)lane * 64, 0, 1);
    }

    v16h afr;                                      // A: attn rows (f32->f16)
    int M = ln16;
    if (M < NSLOT) {
      const float* ar = attb + (size_t)M * NPIX + pk;
      f4 x0 = *(const f4*)(ar + hi * 8);
      f4 x1 = *(const f4*)(ar + hi * 8 + 4);
      f4 x2 = *(const f4*)(ar + 16 + hi * 8);
      f4 x3 = *(const f4*)(ar + 16 + hi * 8 + 4);
      afr[0]=(_Float16)x0.x; afr[1]=(_Float16)x0.y; afr[2]=(_Float16)x0.z; afr[3]=(_Float16)x0.w;
      afr[4]=(_Float16)x1.x; afr[5]=(_Float16)x1.y; afr[6]=(_Float16)x1.z; afr[7]=(_Float16)x1.w;
      afr[8]=(_Float16)x2.x; afr[9]=(_Float16)x2.y; afr[10]=(_Float16)x2.z; afr[11]=(_Float16)x2.w;
      afr[12]=(_Float16)x3.x; afr[13]=(_Float16)x3.y; afr[14]=(_Float16)x3.z; afr[15]=(_Float16)x3.w;
    } else {
#pragma unroll
      for (int e = 0; e < 16; ++e) afr[e] = (_Float16)0.f;
    }
    // wait for the async tile, then feed B fragments from LDS
    asm volatile("s_wait_asynccnt 0x0" ::: "memory");
#pragma unroll
    for (int t = 0; t < 4; ++t) {
      v16h bfr;
      int d = t * 16 + ln16;
#pragma unroll
      for (int e = 0; e < 16; ++e) bfr[e] = vt[w][((e < 8 ? hi * 8 + e : 16 + hi * 8 + (e - 8))) * DIMD + d];
      acc[t] = wmma_f16(afr, bfr, acc[t]);
    }
  }
#pragma unroll
  for (int t = 0; t < 4; ++t)
    if (hi == 0)
#pragma unroll
      for (int r = 0; r < NSLOT; ++r) up[w][r][t * 16 + ln16] = acc[t][r];
  __syncthreads();

  // deterministic wave reduction + normalize by attn row-sum
  for (int idx = tid; idx < NSLOT * DIMD; idx += 256) {
    int i = idx >> 6, d = idx & 63;
    float s = 0.f;
    for (int ww = 0; ww < 8; ++ww) s += up[ww][i][d];
    float rs = 0.f;
    for (int nb = 0; nb < 32; ++nb) rs += spart[((size_t)b * 32 + nb) * 8 + i];
    updf[i][d] = s / rs;
    prevs[i][d] = slots[(size_t)(b * NSLOT + i) * DIMD + d];
    relv[i][d] = 0.f;
  }
  __syncthreads();

  // ---- phase 2: pairwise relational MLP on prev slots -------------------
  for (int p = 0; p < NSLOT * NSLOT; ++p) {
    int i = p / NSLOT, j = p % NSLOT;
    if (tid < HIDD) {
      float a = rb1[tid];
      for (int k = 0; k < DIMD; ++k)
        a += prevs[i][k] * rW1[k * HIDD + tid] + prevs[j][k] * rW1[(DIMD + k) * HIDD + tid];
      hidv[tid] = fmaxf(a, 0.f);
    }
    __syncthreads();
    if (i != j && tid < DIMD) {
      float o = rb2[tid];
      for (int h = 0; h < HIDD; ++h) o += hidv[h] * rW2[h * DIMD + tid];
      relv[i][tid] += o;
    }
    __syncthreads();
  }
  for (int idx = tid; idx < NSLOT * DIMD; idx += 256)
    relv[idx >> 6][idx & 63] *= (1.f / (NSLOT - 1));
  __syncthreads();

  // ---- phase 3: GRU cell ------------------------------------------------
  for (int idx = tid; idx < NSLOT * 3 * DIMD; idx += 256) {
    int s = idx / (3 * DIMD), gx = idx % (3 * DIMD);
    float a = gbih[gx], h2 = gbhh[gx];
    for (int k = 0; k < DIMD; ++k) {
      a  += updf[s][k] * gWih[k * (3 * DIMD) + gx] + relv[s][k] * gWih[(DIMD + k) * (3 * DIMD) + gx];
      h2 += prevs[s][k] * gWhh[k * (3 * DIMD) + gx];
    }
    giA[s][gx] = a; ghA[s][gx] = h2;
  }
  __syncthreads();
  for (int idx = tid; idx < NSLOT * DIMD; idx += 256) {
    int s = idx >> 6, d = idx & 63;
    float rg = 1.f / (1.f + __expf(-(giA[s][d] + ghA[s][d])));
    float zg = 1.f / (1.f + __expf(-(giA[s][DIMD + d] + ghA[s][DIMD + d])));
    float ng = tanhf(giA[s][2 * DIMD + d] + rg * ghA[s][2 * DIMD + d]);
    snew[s][d] = (1.f - zg) * ng + zg * prevs[s][d];
  }
  __syncthreads();

  // ---- phase 4: residual MLP -------------------------------------------
  if (tid < NSLOT) {
    float s = 0.f, s2 = 0.f;
    for (int k = 0; k < DIMD; ++k) { float v2 = snew[tid][k]; s += v2; s2 += v2 * v2; }
    float mean = s / DIMD, var = s2 / DIMD - mean * mean;
    mrow[tid] = mean; ivrow[tid] = rsqrtf(var + LN_EPSF);
  }
  __syncthreads();
  for (int s = 0; s < NSLOT; ++s) {
    if (tid < HIDD) {
      float a = ob1[tid];
      for (int k = 0; k < DIMD; ++k) {
        float hn = (snew[s][k] - mrow[s]) * ivrow[s] * lng[k] + lnb[k];
        a += hn * oW1[k * HIDD + tid];
      }
      hidv[tid] = fmaxf(a, 0.f);
    }
    __syncthreads();
    if (tid < DIMD) {
      float o = ob2[tid];
      for (int h = 0; h < HIDD; ++h) o += hidv[h] * oW2[h * DIMD + tid];
      float res = snew[s][tid] + o;
      slots[(size_t)(b * NSLOT + s) * DIMD + tid] = res;
      if (it == NITER - 1) out_slots[(size_t)(b * NSLOT + s) * DIMD + tid] = res;
    }
    __syncthreads();
  }
}

extern "C" void kernel_launch(void* const* d_in, const int* in_sizes, int n_in,
                              void* d_out, int out_size, void* d_ws, size_t ws_size,
                              hipStream_t stream) {
  (void)in_sizes; (void)n_in; (void)out_size; (void)ws_size;
  const float* x     = (const float*)d_in[0];
  const float* noise = (const float*)d_in[1];
  const float* mu    = (const float*)d_in[2];
  const float* sigma = (const float*)d_in[3];
  const float* Wq    = (const float*)d_in[4];
  const float* Wk    = (const float*)d_in[5];
  const float* Wv    = (const float*)d_in[6];
  const float* gWih  = (const float*)d_in[7];
  const float* gWhh  = (const float*)d_in[8];
  const float* gbih  = (const float*)d_in[9];
  const float* gbhh  = (const float*)d_in[10];
  const float* oW1   = (const float*)d_in[11];
  const float* ob1   = (const float*)d_in[12];
  const float* oW2   = (const float*)d_in[13];
  const float* ob2   = (const float*)d_in[14];
  const float* rW1   = (const float*)d_in[15];
  const float* rb1   = (const float*)d_in[16];
  const float* rW2   = (const float*)d_in[17];
  const float* rb2   = (const float*)d_in[18];
  const float* ling  = (const float*)d_in[19];
  const float* linb  = (const float*)d_in[20];
  const float* lsg   = (const float*)d_in[21];
  const float* lsb   = (const float*)d_in[22];
  const float* lfg   = (const float*)d_in[23];
  const float* lfb   = (const float*)d_in[24];

  size_t KV = (size_t)BBATCH * NPIX * DIMD;
  _Float16* kf = (_Float16*)d_ws;
  _Float16* vf = kf + KV;
  _Float16* qf = vf + KV;
  float* slots = (float*)(qf + (size_t)BBATCH * 16 * DIMD);
  float* spart = slots + BBATCH * NSLOT * DIMD;

  float* outp = (float*)d_out;
  float* attn_out = outp + BBATCH * NSLOT * DIMD;

  k_slot_init<<<(BBATCH * NSLOT * DIMD + 255) / 256, 256, 0, stream>>>(mu, sigma, noise, slots);
  k_ln_kv<<<(BBATCH * NPIX) / 128, 256, 0, stream>>>(x, Wk, Wv, ling, linb, kf, vf);
  for (int it = 0; it < NITER; ++it) {
    k_qproj<<<BBATCH, 256, 0, stream>>>(slots, Wq, lsg, lsb, qf);
    k_dots<<<BBATCH * 32, 256, 0, stream>>>(kf, qf, attn_out, spart, it);
    k_update<<<BBATCH, 256, 0, stream>>>(vf, attn_out, spart, slots,
        gWih, gWhh, gbih, gbhh, oW1, ob1, oW2, ob2, rW1, rb1, rW2, rb2,
        lfg, lfb, outp, it);
  }
}
